// HierarchicalMemoryBank_88605175316807
// MI455X (gfx1250) — matrix operand order repaired and checked by reading, not compiled, not run on hardware
//
#include <hip/hip_runtime.h>

typedef unsigned short u16;
typedef unsigned int   u32;
typedef __attribute__((ext_vector_type(16))) __bf16 v16bf;
typedef __attribute__((ext_vector_type(8)))  float  v8f;

struct alignas(16) U4 { u32 x, y, z, w; };
struct alignas(16) F4 { float x, y, z, w; };
struct alignas(8)  H4 { u16 a, b, c, d; };
union BFrag { v16bf v; U4 q[2]; };

#define BB   32768
#define DD   1024
#define LATD 64
#define SSL  32
#define LSL  256
#define KS   4
#define KL   8

// -------- helpers --------
__device__ __forceinline__ u16 f2bf(float f){
  u32 u = __float_as_uint(f);
  u32 r = u + 0x7FFFu + ((u >> 16) & 1u);   // round-to-nearest-even
  return (u16)(r >> 16);
}
__device__ __forceinline__ float bf2f(u16 h){ return __uint_as_float(((u32)h) << 16); }

__device__ __forceinline__ float wave_max(float v){
  #pragma unroll
  for (int o = 16; o > 0; o >>= 1) v = fmaxf(v, __shfl_xor(v, o, 32));
  return v;
}
__device__ __forceinline__ float wave_sum(float v){
  #pragma unroll
  for (int o = 16; o > 0; o >>= 1) v += __shfl_xor(v, o, 32);
  return v;
}
__device__ __forceinline__ void wave_argmax(float& bv, int& bi){
  #pragma unroll
  for (int o = 16; o > 0; o >>= 1){
    float ov = __shfl_xor(bv, o, 32);
    int   oi = __shfl_xor(bi, o, 32);
    if (ov > bv || (ov == bv && oi < bi)){ bv = ov; bi = oi; }
  }
}

// ================= prep kernels (tiny, run once per launch) =================
// WlT[n][k] = sum_l q_w[k][l] * long_bank[n][l]   (256 x 1024, split bf16)
__global__ void __launch_bounds__(256)
k_prep_wl(const float* __restrict__ q_w, const float* __restrict__ lbank,
          u16* __restrict__ hi, u16* __restrict__ lo){
  int id = blockIdx.x * 256 + threadIdx.x;      // 0 .. 262143
  int n = id >> 10, k = id & 1023;
  const float* a = q_w + (size_t)k * LATD;
  const float* b = lbank + (size_t)n * LATD;
  float s = 0.f;
  #pragma unroll 8
  for (int l = 0; l < LATD; ++l) s = fmaf(a[l], b[l], s);
  u16 h = f2bf(s);
  hi[id] = h;
  lo[id] = f2bf(s - bf2f(h));
}

// WsT[n][k] = sum_d sq_w[k][d] * short_bank[n][d]   (32 x 1024, split bf16)
__global__ void __launch_bounds__(256)
k_prep_ws(const float* __restrict__ sq_w, const float* __restrict__ sbank,
          u16* __restrict__ hi, u16* __restrict__ lo){
  int id = blockIdx.x * 256 + threadIdx.x;      // 0 .. 32767
  int n = id >> 10, k = id & 1023;
  const float* a = sq_w + (size_t)k * DD;
  const float* b = sbank + (size_t)n * DD;
  float s = 0.f;
  #pragma unroll 8
  for (int d = 0; d < DD; ++d) s = fmaf(a[d], b[d], s);
  u16 h = f2bf(s);
  hi[id] = h;
  lo[id] = f2bf(s - bf2f(h));
}

// dec_full[n][d] = sum_l long_bank[n][l]*dec_w[l][d] + dec_b[d]
__global__ void __launch_bounds__(256)
k_prep_dec(const float* __restrict__ lbank, const float* __restrict__ dec_w,
           const float* __restrict__ dec_b, float* __restrict__ decf){
  int id = blockIdx.x * 256 + threadIdx.x;      // 0 .. 262143
  int n = id >> 10, d = id & 1023;
  float s = dec_b[d];
  const float* b = lbank + (size_t)n * LATD;
  #pragma unroll 8
  for (int l = 0; l < LATD; ++l) s = fmaf(b[l], dec_w[(size_t)l * DD + d], s);
  decf[id] = s;
}

// b_l[n] = q_b . long_bank[n];  b_s[n] = sq_b . short_bank[n]
__global__ void __launch_bounds__(256)
k_prep_bias(const float* __restrict__ q_b, const float* __restrict__ lbank,
            const float* __restrict__ sq_b, const float* __restrict__ sbank,
            float* __restrict__ b_l, float* __restrict__ b_s){
  int t = threadIdx.x;
  float s = 0.f;
  for (int l = 0; l < LATD; ++l) s = fmaf(q_b[l], lbank[(size_t)t * LATD + l], s);
  b_l[t] = s;
  if (t < SSL){
    float s2 = 0.f;
    for (int d = 0; d < DD; ++d) s2 = fmaf(sq_b[d], sbank[(size_t)t * DD + d], s2);
    b_s[t] = s2;
  }
}

// ================= fused main kernel =================
struct SmemA { u16 Ahi[16 * DD]; u16 Alo[16 * DD]; };       // 64 KB
struct SmemL { float L[16][LSL]; float S[16][SSL]; };       // 18.4 KB (aliased)
union  Smem  { SmemA a; SmemL l; };

__device__ __forceinline__ v8f wmma3(v16bf ahi, v16bf alo,
                                     const u16* __restrict__ Whi,
                                     const u16* __restrict__ Wlo,
                                     int off, v8f acc){
  BFrag bhi, blo;
  bhi.q[0] = *(const U4*)(Whi + off);
  bhi.q[1] = *(const U4*)(Whi + off + 8);
  blo.q[0] = *(const U4*)(Wlo + off);
  blo.q[1] = *(const U4*)(Wlo + off + 8);
  acc = __builtin_amdgcn_wmma_f32_16x16x32_bf16(false, ahi, false, bhi.v, (short)0, acc, false, false);
  acc = __builtin_amdgcn_wmma_f32_16x16x32_bf16(false, alo, false, bhi.v, (short)0, acc, false, false);
  acc = __builtin_amdgcn_wmma_f32_16x16x32_bf16(false, ahi, false, blo.v, (short)0, acc, false, false);
  return acc;
}

__global__ void __launch_bounds__(256)
k_main(const float* __restrict__ query, const float* __restrict__ sbank,
       const u16* __restrict__ WlThi, const u16* __restrict__ WlTlo,
       const u16* __restrict__ WsThi, const u16* __restrict__ WsTlo,
       const float* __restrict__ b_l, const float* __restrict__ b_s,
       const float* __restrict__ decf,
       float* __restrict__ o_ret, int* __restrict__ o_sidx, int* __restrict__ o_lidx,
       float* __restrict__ o_stop, float* __restrict__ o_ltop)
{
  __shared__ Smem sh;
  __shared__ float s_top_sh[16][KS];
  __shared__ int   s_idx_sh[16][KS];
  __shared__ float l_top_sh[16][KL];
  __shared__ int   l_idx_sh[16][KL];

  const int tid  = threadIdx.x;
  const int wave = tid >> 5, lane = tid & 31;
  const int rowbase = blockIdx.x * 16;

  // ---- stage 16 x 1024 query tile into LDS as split bf16 (float4 loads) ----
  #pragma unroll 4
  for (int i = 0; i < 16; ++i){
    int e = (i * 256 + tid) * 4;                       // element index, 16B aligned
    F4 v = *(const F4*)(query + (size_t)rowbase * DD + e);
    H4 hh, ll;
    hh.a = f2bf(v.x); ll.a = f2bf(v.x - bf2f(hh.a));
    hh.b = f2bf(v.y); ll.b = f2bf(v.y - bf2f(hh.b));
    hh.c = f2bf(v.z); ll.c = f2bf(v.z - bf2f(hh.c));
    hh.d = f2bf(v.w); ll.d = f2bf(v.w - bf2f(hh.d));
    *(H4*)(&sh.a.Ahi[e]) = hh;
    *(H4*)(&sh.a.Alo[e]) = ll;
  }
  __syncthreads();

  const int N   = lane & 15;
  const int kg  = (lane >> 4) * 16;   // A fragment K sub-offset
  const int g16 = (lane >> 4) * 16;   // B fragment K sub-offset
  const u16* Ah = &sh.a.Ahi[(lane & 15) * DD + kg];   // row M = lane & 15
  const u16* Al = &sh.a.Alo[(lane & 15) * DD + kg];

  const u16* bl0h = WlThi + (size_t)(wave * 32 + N) * DD + g16;
  const u16* bl0l = WlTlo + (size_t)(wave * 32 + N) * DD + g16;
  const u16* bl1h = bl0h + (size_t)16 * DD;
  const u16* bl1l = bl0l + (size_t)16 * DD;
  const int  srow = (wave < 2) ? (wave * 16 + N) : 0;
  const u16* bsh  = WsThi + (size_t)srow * DD + g16;
  const u16* bsl  = WsTlo + (size_t)srow * DD + g16;

  v8f accL0 = {}, accL1 = {}, accS = {};
  for (int kb = 0; kb < DD; kb += 32){
    BFrag ahi, alo;
    ahi.q[0] = *(const U4*)(Ah + kb);
    ahi.q[1] = *(const U4*)(Ah + kb + 8);
    alo.q[0] = *(const U4*)(Al + kb);
    alo.q[1] = *(const U4*)(Al + kb + 8);
    accL0 = wmma3(ahi.v, alo.v, bl0h, bl0l, kb, accL0);
    accL1 = wmma3(ahi.v, alo.v, bl1h, bl1l, kb, accL1);
    if (wave < 2) accS = wmma3(ahi.v, alo.v, bsh, bsl, kb, accS);
  }
  __syncthreads();   // everyone done reading A region before aliasing it with logits

  // ---- spill logits (C layout: VGPR v -> M = v + 8*(lane>=16), N = lane&15) ----
  #pragma unroll
  for (int v = 0; v < 8; ++v){
    int m  = (lane >> 4) * 8 + v;
    int n0 = wave * 32 + N;
    sh.l.L[m][n0]      = (accL0[v] + b_l[n0])      * 0.125f;    // 1/sqrt(64)
    sh.l.L[m][n0 + 16] = (accL1[v] + b_l[n0 + 16]) * 0.125f;
    if (wave < 2){
      int ns = wave * 16 + N;
      sh.l.S[m][ns] = (accS[v] + b_s[ns]) * 0.03125f;           // 1/sqrt(1024)
    }
  }
  __syncthreads();

  // ---- softmax + top-k: wave handles rows 2*wave, 2*wave+1 ----
  #pragma unroll
  for (int rr = 0; rr < 2; ++rr){
    int r = wave * 2 + rr;
    int grow = rowbase + r;

    // long: 256 logits, top-8
    float v[8];
    #pragma unroll
    for (int i = 0; i < 8; ++i) v[i] = sh.l.L[r][lane + 32 * i];
    float mx = v[0];
    #pragma unroll
    for (int i = 1; i < 8; ++i) mx = fmaxf(mx, v[i]);
    mx = wave_max(mx);
    float se = 0.f;
    #pragma unroll
    for (int i = 0; i < 8; ++i) se += __expf(v[i] - mx);
    se = wave_sum(se);
    float inv = 1.0f / se;
    #pragma unroll
    for (int t = 0; t < KL; ++t){
      float bv = v[0]; int bi = lane;
      #pragma unroll
      for (int i = 1; i < 8; ++i) if (v[i] > bv){ bv = v[i]; bi = lane + 32 * i; }
      wave_argmax(bv, bi);
      if (lane == 0){
        float p = __expf(bv - mx) * inv;
        o_ltop[(size_t)grow * KL + t] = p;
        o_lidx[(size_t)grow * KL + t] = bi;
        l_top_sh[r][t] = p; l_idx_sh[r][t] = bi;
      }
      #pragma unroll
      for (int i = 0; i < 8; ++i)              // no dynamic register indexing
        if (lane + 32 * i == bi) v[i] = -3.0e38f;
    }

    // short: 32 logits, top-4
    float sv = sh.l.S[r][lane];
    float smx = wave_max(sv);
    float ssum = wave_sum(__expf(sv - smx));
    float sinv = 1.0f / ssum;
    #pragma unroll
    for (int t = 0; t < KS; ++t){
      float bv = sv; int bi = lane;
      wave_argmax(bv, bi);
      if (lane == 0){
        float p = __expf(bv - smx) * sinv;
        o_stop[(size_t)grow * KS + t] = p;
        o_sidx[(size_t)grow * KS + t] = bi;
        s_top_sh[r][t] = p; s_idx_sh[r][t] = bi;
      }
      if (lane == bi) sv = -3.0e38f;
    }
  }
  __syncthreads();

  // ---- retrieval: 16 threads per row, float4 gathers from L2-resident tables ----
  const int m = tid >> 4, c = tid & 15;
  float swt[KS]; int six[KS];
  float lwt[KL]; int lix[KL];
  #pragma unroll
  for (int k = 0; k < KS; ++k){ swt[k] = s_top_sh[m][k]; six[k] = s_idx_sh[m][k]; }
  #pragma unroll
  for (int k = 0; k < KL; ++k){ lwt[k] = l_top_sh[m][k]; lix[k] = l_idx_sh[m][k]; }
  const size_t orow = (size_t)(rowbase + m) * DD;
  #pragma unroll 4
  for (int i = 0; i < 16; ++i){
    int d = c * 4 + i * 64;
    F4 acc = {0.f, 0.f, 0.f, 0.f};
    #pragma unroll
    for (int k = 0; k < KS; ++k){
      F4 b = *(const F4*)(sbank + (size_t)six[k] * DD + d);
      acc.x = fmaf(swt[k], b.x, acc.x); acc.y = fmaf(swt[k], b.y, acc.y);
      acc.z = fmaf(swt[k], b.z, acc.z); acc.w = fmaf(swt[k], b.w, acc.w);
    }
    #pragma unroll
    for (int k = 0; k < KL; ++k){
      F4 b = *(const F4*)(decf + (size_t)lix[k] * DD + d);
      acc.x = fmaf(lwt[k], b.x, acc.x); acc.y = fmaf(lwt[k], b.y, acc.y);
      acc.z = fmaf(lwt[k], b.z, acc.z); acc.w = fmaf(lwt[k], b.w, acc.w);
    }
    *(F4*)(o_ret + orow + d) = acc;
  }
}

// ================= launcher =================
extern "C" void kernel_launch(void* const* d_in, const int* in_sizes, int n_in,
                              void* d_out, int out_size, void* d_ws, size_t ws_size,
                              hipStream_t stream){
  (void)in_sizes; (void)n_in; (void)out_size; (void)ws_size;
  const float* query = (const float*)d_in[0];
  const float* sbank = (const float*)d_in[1];
  const float* lbank = (const float*)d_in[2];
  const float* dec_w = (const float*)d_in[3];
  const float* dec_b = (const float*)d_in[4];
  const float* q_w   = (const float*)d_in[5];
  const float* q_b   = (const float*)d_in[6];
  const float* sq_w  = (const float*)d_in[7];
  const float* sq_b  = (const float*)d_in[8];

  unsigned char* ws = (unsigned char*)d_ws;
  u16*   wlhi = (u16*)  (ws + 0);          // 256*1024*2 = 512 KB
  u16*   wllo = (u16*)  (ws + 524288);     // 512 KB
  u16*   wshi = (u16*)  (ws + 1048576);    // 64 KB
  u16*   wslo = (u16*)  (ws + 1114112);    // 64 KB
  float* decf = (float*)(ws + 1179648);    // 1 MB
  float* bl   = (float*)(ws + 2228224);    // 1 KB
  float* bs   = (float*)(ws + 2229248);    // 128 B

  float* out    = (float*)d_out;
  float* o_ret  = out;                                  // (B, 1024) f32
  int*   o_sidx = (int*)(out + (size_t)BB * DD);        // (B, 4)  i32
  int*   o_lidx = o_sidx + (size_t)BB * KS;             // (B, 8)  i32
  float* o_stop = (float*)(o_lidx + (size_t)BB * KL);   // (B, 4)  f32
  float* o_ltop = o_stop + (size_t)BB * KS;             // (B, 8)  f32

  k_prep_wl  <<<1024, 256, 0, stream>>>(q_w, lbank, wlhi, wllo);
  k_prep_ws  <<<128,  256, 0, stream>>>(sq_w, sbank, wshi, wslo);
  k_prep_dec <<<1024, 256, 0, stream>>>(lbank, dec_w, dec_b, decf);
  k_prep_bias<<<1,    256, 0, stream>>>(q_b, lbank, sq_b, sbank, bl, bs);
  k_main     <<<BB / 16, 256, 0, stream>>>(query, sbank, wlhi, wllo, wshi, wslo,
                                           bl, bs, decf,
                                           o_ret, o_sidx, o_lidx, o_stop, o_ltop);
}
